// BahdanauAttention_18622978195571
// MI455X (gfx1250) — matrix operand order repaired and checked by reading, not compiled, not run on hardware
//
#include <hip/hip_runtime.h>
#include <hip/hip_bf16.h>

typedef __attribute__((ext_vector_type(16))) __bf16 v16bf;
typedef __attribute__((ext_vector_type(4)))  __bf16 v4bf;
typedef __attribute__((ext_vector_type(8)))  float  v8f;
typedef __attribute__((ext_vector_type(4)))  float  vf4;
typedef __attribute__((ext_vector_type(4)))  int    vi4;

typedef __attribute__((address_space(1))) vi4 gvi4;   // global int4
typedef __attribute__((address_space(3))) vi4 lvi4;   // LDS int4

#if __has_builtin(__builtin_amdgcn_global_load_async_to_lds_b128) && \
    __has_builtin(__builtin_amdgcn_s_wait_asynccnt)
#define USE_ASYNC_LDS 1
#endif

namespace {
constexpr int kU = 512;                 // units
constexpr int kE = 1024;                // 2*units (feature dim of values)
constexpr int kB = 64;                  // batch
constexpr int kS = 2048;                // seq
constexpr int kRM = 32;                 // rows (M) per block
constexpr int kSeg = 8;                 // seq segments for context pass
constexpr int kXStride = kE + 16;       // bf16 elems per LDS X row (2080B stride, conflict-free)
constexpr int kBStride = kU + 4;        // floats per LDS bias row (pad)
}

// ------------------------------------------------------------------
// Kernel P: pack W1 (f32 [512,1024]) into bf16 WMMA B-fragments.
// B (K=32 x N=16, 16-bit), contiguous k order used here (not AMD's table order): lane (n + 16h) holds column n, K = 16h..16h+15.
// Wp[((ut*32 + kt)*32 + lane)*16 + j] = bf16(W1[ut*16 + (lane&15)][kt*32 + 16*(lane>>4) + j])
// ------------------------------------------------------------------
__global__ void pack_w1_kernel(const float* __restrict__ W1, __bf16* __restrict__ Wp) {
  int idx  = blockIdx.x * 256 + threadIdx.x;    // 0 .. 524287
  int j    = idx & 15;
  int lane = (idx >> 4) & 31;
  int kt   = (idx >> 9) & 31;
  int ut   = idx >> 14;
  int n = lane & 15, h = lane >> 4;
  Wp[idx] = (__bf16)W1[(size_t)(ut * 16 + n) * kE + kt * 32 + 16 * h + j];
}

// ------------------------------------------------------------------
// Kernel A: bias[b][u] = W1_b[u] + W2_b[u] + sum_k query[b,k]*W2_w[u,k]
// ------------------------------------------------------------------
__global__ void bias_kernel(const float* __restrict__ q, const float* __restrict__ W2w,
                            const float* __restrict__ W2b, const float* __restrict__ W1b,
                            float* __restrict__ bias) {
  int b = blockIdx.x;
  int u = blockIdx.y * 256 + threadIdx.x;
  const float* qp = q + b * kU;
  const float* wp = W2w + (size_t)u * kU;
  float acc = 0.f;
#pragma unroll 4
  for (int k = 0; k < kU; ++k) acc = fmaf(qp[k], wp[k], acc);
  bias[b * kU + u] = acc + W2b[u] + W1b[u];
}

// ------------------------------------------------------------------
// Kernel B: scores. Block = 32 rows of X (rows r = s*64+b, contiguous).
// 8 waves each own a 32x64 C strip (2 M-tiles x 4 N-tiles) so every B
// fragment feeds 2 WMMAs (halves W1 re-read traffic from L2).
// score[r] = sum_u Vw[u] * tanh((X W1^T)[r,u] + bias[r&63][u])  (V_b dropped:
// softmax-invariant). Scores land in the attn region of d_out at [b*2048+s].
// ------------------------------------------------------------------
__launch_bounds__(256)
__global__ void score_kernel(const float* __restrict__ values,
                             const __bf16* __restrict__ Wp,
                             const float* __restrict__ bias,
                             const float* __restrict__ Vw,
                             float* __restrict__ attnbuf) {
  __shared__ __align__(16) __bf16 lds_x[kRM * kXStride];      // ~66.5 KB
  __shared__ __align__(16) float  lds_bias[kRM * kBStride];   // ~66 KB
  __shared__ __align__(16) float  lds_vw[kU];
  __shared__ float lds_part[8][kRM];

  const int tid  = threadIdx.x;
  const int lane = tid & 31;
  const int wave = tid >> 5;
  const int r0   = blockIdx.x * kRM;    // first row of this tile
  const int c0   = r0 & 63;             // first batch index (rows span b = c0..c0+31)

  // ---- stage X tile (32 x 1024 f32) as bf16 into LDS, coalesced float4 loads
  const float* xb = values + (size_t)r0 * kE;
#pragma unroll
  for (int i = 0; i < 32; ++i) {
    int c   = tid + 256 * i;            // float4 chunk id, 0..8191
    int row = c >> 8;                   // 256 chunks per row
    int col = (c & 255) << 2;
    vf4 f = *(const vf4*)(xb + (size_t)row * kE + col);
    v4bf h4 = { (__bf16)f.x, (__bf16)f.y, (__bf16)f.z, (__bf16)f.w };
    *(v4bf*)(&lds_x[row * kXStride + col]) = h4;   // ds_store_b64
  }

  // ---- stage bias rows c0..c0+31 (32 x 512 f32) + Vw
#ifdef USE_ASYNC_LDS
#pragma unroll
  for (int i = 0; i < 16; ++i) {
    int c   = tid + 256 * i;            // 0..4095
    int row = c >> 7;                   // 128 float4 chunks per row
    int col = (c & 127) << 2;
    __builtin_amdgcn_global_load_async_to_lds_b128(
        (gvi4*)(bias + (size_t)(c0 + row) * kU + col),
        (lvi4*)(&lds_bias[row * kBStride + col]),
        0, 0);
  }
  if (tid < 128)
    __builtin_amdgcn_global_load_async_to_lds_b128(
        (gvi4*)(Vw + tid * 4),
        (lvi4*)(&lds_vw[tid * 4]),
        0, 0);
  __builtin_amdgcn_s_wait_asynccnt(0);
#else
#pragma unroll
  for (int i = 0; i < 16; ++i) {
    int c   = tid + 256 * i;
    int row = c >> 7;
    int col = (c & 127) << 2;
    *(vf4*)(&lds_bias[row * kBStride + col]) =
        *(const vf4*)(bias + (size_t)(c0 + row) * kU + col);
  }
  if (tid < 128) *(vf4*)(&lds_vw[tid * 4]) = *(const vf4*)(Vw + tid * 4);
#endif
  __syncthreads();

  // ---- main GEMM: 2 M-tiles x 4 N-tiles of 16x16, bf16 WMMA
  v8f acc[2][4];
#pragma unroll
  for (int mt = 0; mt < 2; ++mt)
#pragma unroll
    for (int t = 0; t < 4; ++t) acc[mt][t] = v8f{};

  const int g = lane >> 4;              // A-matrix K-group (16-bit layout)
  const int m = lane & 15;              // A row within 16-row tile
  const __bf16* wpW = Wp + (size_t)(wave * 4) * (32 * 32 * 16);  // ut = 4*wave + t

  for (int kt = 0; kt < 32; ++kt) {
    const int k0 = kt * 32;
    // A fragments: A 16x32, contiguous k order used here -> group g holds K=16g..16g+15 (matches packed B)
    union AFrag { vi4 q[2]; v16bf v; } a[2];
#pragma unroll
    for (int mt = 0; mt < 2; ++mt) {
      const int row = mt * 16 + m;
      a[mt].q[0] = *(const vi4*)(&lds_x[row * kXStride + k0 + 16 * g]);      // ds_load_b128
      a[mt].q[1] = *(const vi4*)(&lds_x[row * kXStride + k0 + 16 * g + 8]);  // ds_load_b128
    }
#pragma unroll
    for (int t = 0; t < 4; ++t) {
      v16bf bf = *(const v16bf*)(wpW + t * 16384 + kt * 512 + lane * 16);
      acc[0][t] = __builtin_amdgcn_wmma_f32_16x16x32_bf16(
          false, a[0].v, false, bf, (short)0, acc[0][t], false, false);
      acc[1][t] = __builtin_amdgcn_wmma_f32_16x16x32_bf16(
          false, a[1].v, false, bf, (short)0, acc[1][t], false, false);
    }
  }

  // ---- epilogue: tanh + dot with Vw.
  // C layout: element v in lane L -> row M = v + 8*(L>>4), col N = (L&15).
  const int n   = lane & 15;
  const int mhi = (lane >> 4) * 8;
  float p[2][8];
#pragma unroll
  for (int mt = 0; mt < 2; ++mt)
#pragma unroll
    for (int v = 0; v < 8; ++v) p[mt][v] = 0.f;
#pragma unroll
  for (int t = 0; t < 4; ++t) {
    const int uu = wave * 64 + t * 16 + n;
    const float vwu = lds_vw[uu];
#pragma unroll
    for (int mt = 0; mt < 2; ++mt)
#pragma unroll
      for (int v = 0; v < 8; ++v) {
        float x = acc[mt][t][v] + lds_bias[(mt * 16 + mhi + v) * kBStride + uu];
        p[mt][v] += tanhf(x) * vwu;
      }
  }
  // deterministic reduction: shfl within each 16-lane group, fixed-order LDS sum
#pragma unroll
  for (int mt = 0; mt < 2; ++mt)
#pragma unroll
    for (int v = 0; v < 8; ++v) {
      float x = p[mt][v];
#pragma unroll
      for (int off = 8; off >= 1; off >>= 1) x += __shfl_xor(x, off, 16);
      if (n == 0) lds_part[wave][mt * 16 + mhi + v] = x;
    }
  __syncthreads();
  if (tid < kRM) {
    float s = 0.f;
#pragma unroll
    for (int w = 0; w < 8; ++w) s += lds_part[w][tid];
    const int r = r0 + tid;
    attnbuf[(size_t)(r & 63) * kS + (r >> 6)] = s;   // [b*2048 + s]
  }
}

// ------------------------------------------------------------------
// Kernel C: softmax over seq per batch, in place in the attn region.
// ------------------------------------------------------------------
__global__ void softmax_kernel(float* __restrict__ attn) {
  __shared__ float red[8];
  const int b = blockIdx.x, tid = threadIdx.x;
  const int lane = tid & 31, wv = tid >> 5;
  float* base = attn + (size_t)b * kS;
  float v[8];
  float mx = -3.402823466e38f;
#pragma unroll
  for (int i = 0; i < 8; ++i) { v[i] = base[tid + 256 * i]; mx = fmaxf(mx, v[i]); }
#pragma unroll
  for (int off = 16; off >= 1; off >>= 1) mx = fmaxf(mx, __shfl_xor(mx, off, 32));
  if (lane == 0) red[wv] = mx;
  __syncthreads();
  float gmx = red[0];
#pragma unroll
  for (int w = 1; w < 8; ++w) gmx = fmaxf(gmx, red[w]);
  __syncthreads();
  float sum = 0.f;
#pragma unroll
  for (int i = 0; i < 8; ++i) { v[i] = __expf(v[i] - gmx); sum += v[i]; }
#pragma unroll
  for (int off = 16; off >= 1; off >>= 1) sum += __shfl_xor(sum, off, 32);
  if (lane == 0) red[wv] = sum;
  __syncthreads();
  float gs = 0.f;
#pragma unroll
  for (int w = 0; w < 8; ++w) gs += red[w];
  const float inv = 1.0f / gs;
#pragma unroll
  for (int i = 0; i < 8; ++i) base[tid + 256 * i] = v[i] * inv;
}

// ------------------------------------------------------------------
// Kernel D1: partial context over a seq segment.
// part[b][seg][e] = sum_{s in seg} attn[b][s] * values[s][b][e]
// ------------------------------------------------------------------
__global__ void context_part_kernel(const float* __restrict__ values,
                                    const float* __restrict__ attn,
                                    float* __restrict__ part) {
  const int b   = blockIdx.x;
  const int e   = blockIdx.y * 256 + threadIdx.x;
  const int seg = blockIdx.z;
  const int s0  = seg * (kS / kSeg);
  const float* vp = values + (size_t)b * kE + e;
  const float* ap = attn + (size_t)b * kS;
  float acc = 0.f;
  for (int s = s0; s < s0 + kS / kSeg; s += 4) {
    __builtin_prefetch(vp + (size_t)(s + 16) * (kB * kE), 0, 1);  // global_prefetch_b8
    const float a0 = ap[s], a1 = ap[s + 1], a2 = ap[s + 2], a3 = ap[s + 3];
    acc = fmaf(a0, vp[(size_t)(s + 0) * (kB * kE)], acc);
    acc = fmaf(a1, vp[(size_t)(s + 1) * (kB * kE)], acc);
    acc = fmaf(a2, vp[(size_t)(s + 2) * (kB * kE)], acc);
    acc = fmaf(a3, vp[(size_t)(s + 3) * (kB * kE)], acc);
  }
  part[((size_t)b * kSeg + seg) * kE + e] = acc;
}

// ------------------------------------------------------------------
// Kernel D2: fixed-order (deterministic) segment reduction.
// ------------------------------------------------------------------
__global__ void context_finish_kernel(const float* __restrict__ part,
                                      float* __restrict__ ctx) {
  const int idx = blockIdx.x * 256 + threadIdx.x;  // 0..65535
  const int b = idx >> 10, e = idx & 1023;
  float s = 0.f;
#pragma unroll
  for (int g = 0; g < kSeg; ++g) s += part[((size_t)b * kSeg + g) * kE + e];
  ctx[idx] = s;
}

// ------------------------------------------------------------------
extern "C" void kernel_launch(void* const* d_in, const int* in_sizes, int n_in,
                              void* d_out, int out_size, void* d_ws, size_t ws_size,
                              hipStream_t stream) {
  const float* query  = (const float*)d_in[0];   // [64,512]
  const float* values = (const float*)d_in[1];   // [2048,64,1024]
  const float* W1w    = (const float*)d_in[2];   // [512,1024]
  const float* W1b    = (const float*)d_in[3];   // [512]
  const float* W2w    = (const float*)d_in[4];   // [512,512]
  const float* W2b    = (const float*)d_in[5];   // [512]
  const float* Vw     = (const float*)d_in[6];   // [1,512]
  // d_in[7] = V_b: constant shift of scores, softmax-invariant -> unused.

  float* out  = (float*)d_out;
  float* ctx  = out;               // [64,1024] context
  float* attn = out + kB * kE;     // [64,2048] attn (scores live here pre-softmax)

  char*   ws   = (char*)d_ws;
  float*  bias = (float*)ws;                      // 64*512 f32        = 128 KB
  __bf16* Wp   = (__bf16*)(ws + (128 << 10));     // 512K bf16         = 1 MB
  float*  part = (float*)(ws + (1152 << 10));     // 64*8*1024 f32     = 2 MB

  pack_w1_kernel<<<2048, 256, 0, stream>>>(W1w, Wp);
  bias_kernel<<<dim3(kB, kU / 256), 256, 0, stream>>>(query, W2w, W2b, W1b, bias);
  score_kernel<<<(kS * kB) / kRM, 256, 0, stream>>>(values, Wp, bias, Vw, attn);
  softmax_kernel<<<kB, 256, 0, stream>>>(attn);
  context_part_kernel<<<dim3(kB, kE / 256, kSeg), 256, 0, stream>>>(values, attn, part);
  context_finish_kernel<<<(kB * kE) / 256, 256, 0, stream>>>(part, ctx);
}
